// MyEncoderLayer_6219112644792
// MI455X (gfx1250) — compile-verified, matrix operand order checked by reading, not run on hardware
//
#include <hip/hip_runtime.h>
#include <math.h>
#include <stdint.h>

// ---------------------------------------------------------------------------
// Types for CDNA5 WMMA (wave32): V_WMMA_F32_16X16X32_BF16
// ---------------------------------------------------------------------------
typedef __bf16 bf16;
typedef __attribute__((ext_vector_type(16))) __bf16 bf16x16;
typedef __attribute__((ext_vector_type(8)))  __bf16 bf16x8;
typedef __attribute__((ext_vector_type(8)))  float  f32x8;

#define HID   1024
#define NSEQ  4096
#define BATCH 8
#define HEADS 16
#define DH    64
#define ROWS  (BATCH * NSEQ)   // 32768
#define BH    (BATCH * HEADS)  // 128

__device__ __forceinline__ f32x8 wmma_bf16(bf16x16 a, bf16x16 b, f32x8 c) {
  // 8 args: (neg_a, A, neg_b, B, c_mod, C, reuse_a, reuse_b)
  return __builtin_amdgcn_wmma_f32_16x16x32_bf16(false, a, false, b, (short)0, c,
                                                 false, false);
}

// CDNA5 async copy: global -> LDS, 16 bytes, tracked by ASYNCcnt.
// lds_off = wave-relative LDS byte offset (low 32 bits of generic LDS pointer,
// per ISA 10.2 aperture mapping). gaddr = 64-bit global address.
__device__ __forceinline__ void async_cp_b128(uint32_t lds_off, const void* gptr) {
  asm volatile("global_load_async_to_lds_b128 %0, %1, off"
               :
               : "v"(lds_off), "v"((unsigned long long)(uintptr_t)gptr)
               : "memory");
}

__device__ __forceinline__ void wait_async0() {
  asm volatile("s_wait_asynccnt 0x0" ::: "memory");
}

__device__ __forceinline__ uint32_t lds_off32(const void* p) {
  return (uint32_t)(uintptr_t)p;
}

// Load one 16x32-K bf16 fragment (A: lane=row, B: lane=col).
// p points at the row/col base (element 0 of the 32-wide K chunk).
// Element e of the vector maps to K = (e<8 ? half*8+e : 16+half*8+(e-8)).
__device__ __forceinline__ bf16x16 frag_ld(const bf16* p, int half) {
  bf16x8 lo = *(const bf16x8*)(p + half * 8);
  bf16x8 hi = *(const bf16x8*)(p + 16 + half * 8);
  bf16x16 r;
#pragma unroll
  for (int i = 0; i < 8; ++i) { r[i] = lo[i]; r[i + 8] = hi[i]; }
  return r;
}

__device__ __forceinline__ f32x8 zero8() {
  f32x8 z = {0.f, 0.f, 0.f, 0.f, 0.f, 0.f, 0.f, 0.f};
  return z;
}

// ---------------------------------------------------------------------------
// Weight transpose + f32->bf16 :  WT[n][k] = (bf16) W[k][n]   (1024x1024)
// ---------------------------------------------------------------------------
__global__ __launch_bounds__(256) void transpose_w(const float* __restrict__ W,
                                                   bf16* __restrict__ WT) {
  __shared__ float tile[32][33];
  const int tx = threadIdx.x & 31;
  const int ty = threadIdx.x >> 5;           // 0..7
  const int j0 = blockIdx.x * 32;            // n tile
  const int k0 = blockIdx.y * 32;            // k tile
#pragma unroll
  for (int r = 0; r < 32; r += 8)
    tile[ty + r][tx] = W[(size_t)(k0 + ty + r) * HID + (j0 + tx)];
  __syncthreads();
#pragma unroll
  for (int r = 0; r < 32; r += 8)
    WT[(size_t)(j0 + ty + r) * HID + (k0 + tx)] = (bf16)tile[tx][ty + r];
}

// x f32 -> bf16
__global__ __launch_bounds__(256) void cvt_x(const float* __restrict__ x,
                                             bf16* __restrict__ xb) {
  size_t i = (size_t)blockIdx.x * 256 + threadIdx.x;
  xb[i] = (bf16)x[i];
}

// ---------------------------------------------------------------------------
// Main WMMA GEMM: out[M=32768, N=1024] = A[M,1024](bf16) * Bt[N,1024](bf16)^T
// Block = 256 threads = 8 waves, tile 128(M) x 64(N); wave = 32x32 (2x2 WMMA)
// Tiles staged into LDS via global_load_async_to_lds_b128 (ASYNCcnt).
// MODE 0: h   -> f32 out = acc + bias
// MODE 1: q   -> bf16 sigmoid(acc) scattered to qH[bh][n][d]
// MODE 2: k   -> bf16 sigmoid(acc) scattered to kH[bh][d][n]
// MODE 3: v   -> bf16 acc           scattered to vH[bh][d][n]
// MODE 4: out -> f32 relu(acc + bias)
// ---------------------------------------------------------------------------
template <int MODE>
__global__ __launch_bounds__(256) void gemm1024(const bf16* __restrict__ A,
                                                const bf16* __restrict__ Bt,
                                                const float* __restrict__ bias,
                                                float* __restrict__ outF,
                                                bf16* __restrict__ outB) {
  __shared__ __align__(16) bf16 As[128 * 40];  // 32-K rows padded to 40
  __shared__ __align__(16) bf16 Bs[64 * 40];

  const int tid   = threadIdx.x;
  const int lane  = tid & 31;
  const int wave  = tid >> 5;
  const int idx   = lane & 15;
  const int half  = lane >> 4;
  const int waveM = wave >> 1;  // 0..3
  const int waveN = wave & 1;   // 0..1
  const int m0    = blockIdx.x * 128;
  const int n0    = blockIdx.y * 64;

  // Per-thread staging chunk coordinates (fixed over the K loop).
  const int rowA0 = (tid * 2) >> 2,     offA0 = (tid * 2) & 3;
  const int rowA1 = (tid * 2 + 1) >> 2, offA1 = (tid * 2 + 1) & 3;
  const int rowB  = tid >> 2,           offB  = tid & 3;
  const uint32_t ldsA0 = lds_off32(&As[rowA0 * 40 + offA0 * 8]);
  const uint32_t ldsA1 = lds_off32(&As[rowA1 * 40 + offA1 * 8]);
  const uint32_t ldsB  = lds_off32(&Bs[rowB * 40 + offB * 8]);

  f32x8 acc[2][2];
  acc[0][0] = zero8(); acc[0][1] = zero8();
  acc[1][0] = zero8(); acc[1][1] = zero8();

  for (int k0 = 0; k0 < HID; k0 += 32) {
    // Async DMA: A tile 128x32 (2 x b128/thread), B tile 64x32 (1 x b128/thread)
    async_cp_b128(ldsA0, &A[(size_t)(m0 + rowA0) * HID + k0 + offA0 * 8]);
    async_cp_b128(ldsA1, &A[(size_t)(m0 + rowA1) * HID + k0 + offA1 * 8]);
    async_cp_b128(ldsB,  &Bt[(size_t)(n0 + rowB) * HID + k0 + offB * 8]);
    wait_async0();
    __syncthreads();

    bf16x16 a0 = frag_ld(&As[(waveM * 32 + idx) * 40], half);
    bf16x16 a1 = frag_ld(&As[(waveM * 32 + 16 + idx) * 40], half);
    bf16x16 b0 = frag_ld(&Bs[(waveN * 32 + idx) * 40], half);
    bf16x16 b1 = frag_ld(&Bs[(waveN * 32 + 16 + idx) * 40], half);

    acc[0][0] = wmma_bf16(a0, b0, acc[0][0]);
    acc[0][1] = wmma_bf16(a0, b1, acc[0][1]);
    acc[1][0] = wmma_bf16(a1, b0, acc[1][0]);
    acc[1][1] = wmma_bf16(a1, b1, acc[1][1]);
    __syncthreads();
  }

  // Epilogue. C layout: row-in-tile = half*8+g, col-in-tile = idx.
#pragma unroll
  for (int i = 0; i < 2; ++i) {
#pragma unroll
    for (int jj = 0; jj < 2; ++jj) {
      const int col = n0 + waveN * 32 + jj * 16 + idx;  // hidden index j
      float bv = 0.f;
      if (MODE == 0 || MODE == 4) bv = bias[col];
#pragma unroll
      for (int g = 0; g < 8; ++g) {
        const int m = m0 + waveM * 32 + i * 16 + half * 8 + g;  // token row
        float v = acc[i][jj][g];
        if (MODE == 0) {
          outF[(size_t)m * HID + col] = v + bv;
        } else if (MODE == 4) {
          outF[(size_t)m * HID + col] = fmaxf(v + bv, 0.f);
        } else {
          const int b  = m >> 12;       // batch
          const int n  = m & 4095;      // seq pos
          const int d  = col >> 4;      // j = d*16 + h
          const int hh = col & 15;
          const size_t bh = (size_t)b * HEADS + hh;
          if (MODE == 1) {  // q, sigmoid -> [bh][n][d]
            float s = 1.f / (1.f + __expf(-v));
            outB[(bh * NSEQ + n) * DH + d] = (bf16)s;
          } else if (MODE == 2) {  // k, sigmoid -> [bh][d][n]
            float s = 1.f / (1.f + __expf(-v));
            outB[(bh * DH + d) * NSEQ + n] = (bf16)s;
          } else {  // v -> [bh][d][n]
            outB[(bh * DH + d) * NSEQ + n] = (bf16)v;
          }
        }
      }
    }
  }
}

// ---------------------------------------------------------------------------
// kv GEMM (split-K=8): kvT[m][d] = sum_n v[n,m]*k[n,d]
//   A = vH[bh] : 64 rows (m) x 4096 (n);  B = kH[bh] : 64 rows (d) x 4096 (n)
// Block = 128 threads (4 waves, 2x2 of 32x32); one block per (slice, bh).
// ---------------------------------------------------------------------------
__global__ __launch_bounds__(128) void kv_gemm(const bf16* __restrict__ vH,
                                               const bf16* __restrict__ kH,
                                               float* __restrict__ kvPart) {
  __shared__ __align__(16) bf16 As[64 * 40];
  __shared__ __align__(16) bf16 Bs[64 * 40];

  const int tid = threadIdx.x, lane = tid & 31, wave = tid >> 5;
  const int idx = lane & 15, half = lane >> 4;
  const int waveM = wave >> 1, waveN = wave & 1;
  const int slice = blockIdx.x;   // 0..7
  const int bh    = blockIdx.y;   // 0..127

  const bf16* Ab = vH + (size_t)bh * DH * NSEQ;
  const bf16* Bb = kH + (size_t)bh * DH * NSEQ;
  const int kbase = slice * 512;

  const int row0 = (tid * 2) >> 2,     off0 = (tid * 2) & 3;
  const int row1 = (tid * 2 + 1) >> 2, off1 = (tid * 2 + 1) & 3;
  const uint32_t ldsA0 = lds_off32(&As[row0 * 40 + off0 * 8]);
  const uint32_t ldsA1 = lds_off32(&As[row1 * 40 + off1 * 8]);
  const uint32_t ldsB0 = lds_off32(&Bs[row0 * 40 + off0 * 8]);
  const uint32_t ldsB1 = lds_off32(&Bs[row1 * 40 + off1 * 8]);

  f32x8 acc[2][2];
  acc[0][0] = zero8(); acc[0][1] = zero8();
  acc[1][0] = zero8(); acc[1][1] = zero8();

  for (int k0 = kbase; k0 < kbase + 512; k0 += 32) {
    async_cp_b128(ldsA0, &Ab[(size_t)row0 * NSEQ + k0 + off0 * 8]);
    async_cp_b128(ldsA1, &Ab[(size_t)row1 * NSEQ + k0 + off1 * 8]);
    async_cp_b128(ldsB0, &Bb[(size_t)row0 * NSEQ + k0 + off0 * 8]);
    async_cp_b128(ldsB1, &Bb[(size_t)row1 * NSEQ + k0 + off1 * 8]);
    wait_async0();
    __syncthreads();

    bf16x16 a0 = frag_ld(&As[(waveM * 32 + idx) * 40], half);
    bf16x16 a1 = frag_ld(&As[(waveM * 32 + 16 + idx) * 40], half);
    bf16x16 b0 = frag_ld(&Bs[(waveN * 32 + idx) * 40], half);
    bf16x16 b1 = frag_ld(&Bs[(waveN * 32 + 16 + idx) * 40], half);

    acc[0][0] = wmma_bf16(a0, b0, acc[0][0]);
    acc[0][1] = wmma_bf16(a0, b1, acc[0][1]);
    acc[1][0] = wmma_bf16(a1, b0, acc[1][0]);
    acc[1][1] = wmma_bf16(a1, b1, acc[1][1]);
    __syncthreads();
  }

  float* out = kvPart + ((size_t)bh * 8 + slice) * (DH * DH);
#pragma unroll
  for (int i = 0; i < 2; ++i)
#pragma unroll
    for (int jj = 0; jj < 2; ++jj)
#pragma unroll
      for (int g = 0; g < 8; ++g) {
        int m = waveM * 32 + i * 16 + half * 8 + g;
        int d = waveN * 32 + jj * 16 + idx;
        out[m * DH + d] = acc[i][jj][g];
      }
}

// Sum split-K partials, convert to bf16: kvT[bh][m][d]
__global__ __launch_bounds__(256) void kv_reduce(const float* __restrict__ kvPart,
                                                 bf16* __restrict__ kvT) {
  int i = blockIdx.x * 256 + threadIdx.x;  // < 128*4096
  int bh = i >> 12, r = i & 4095;
  float s = 0.f;
#pragma unroll
  for (int sl = 0; sl < 8; ++sl)
    s += kvPart[((size_t)bh * 8 + sl) * 4096 + r];
  kvT[i] = (bf16)s;
}

// k_sum[bh*64+d] = sum_n kH[bh][d][n]
__global__ __launch_bounds__(128) void ksum_kernel(const bf16* __restrict__ kH,
                                                   float* __restrict__ ksum) {
  __shared__ float red[128];
  const int row = blockIdx.x;  // bh*64 + d
  const bf16* p = kH + (size_t)row * NSEQ;
  float s = 0.f;
  for (int n = threadIdx.x; n < NSEQ; n += 128) s += (float)p[n];
  red[threadIdx.x] = s;
  __syncthreads();
  for (int o = 64; o > 0; o >>= 1) {
    if (threadIdx.x < o) red[threadIdx.x] += red[threadIdx.x + o];
    __syncthreads();
  }
  if (threadIdx.x == 0) ksum[row] = red[0];
}

// den[bh][n] = dot(q[bh][n][:], ksum[bh][:])
__global__ __launch_bounds__(256) void den_kernel(const bf16* __restrict__ qH,
                                                  const float* __restrict__ ksum,
                                                  float* __restrict__ den) {
  __shared__ float ks[DH];
  const int bh = blockIdx.y;
  if (threadIdx.x < DH) ks[threadIdx.x] = ksum[bh * DH + threadIdx.x];
  __syncthreads();
  const int n = blockIdx.x * 256 + threadIdx.x;
  const bf16* q = qH + ((size_t)bh * NSEQ + n) * DH;
  float s = 0.f;
#pragma unroll
  for (int d = 0; d < DH; ++d) s += (float)q[d] * ks[d];
  den[(size_t)bh * NSEQ + n] = s;
}

// ---------------------------------------------------------------------------
// num GEMM + divide: attn[bh][n][m] = (q[bh] @ kvT[bh]^T)[n][m] / den[bh][n]
//   A = qH[bh] : [n][64] bf16 ; B = kvT[bh] : [m][64] bf16 (K=d contiguous)
// Block = 256 thr = 8 waves (4x2), tile 128(n) x 64(m); K=64 -> 2 steps.
// B is 8KB per head -> L2 resident, no LDS staging needed.
// ---------------------------------------------------------------------------
__global__ __launch_bounds__(256) void num_gemm(const bf16* __restrict__ qH,
                                                const bf16* __restrict__ kvT,
                                                const float* __restrict__ den,
                                                float* __restrict__ attn) {
  const int tid = threadIdx.x, lane = tid & 31, wave = tid >> 5;
  const int idx = lane & 15, half = lane >> 4;
  const int waveM = wave >> 1, waveN = wave & 1;
  const int bh = blockIdx.y;
  const int rowBase = blockIdx.x * 128;

  const bf16* Ab = qH + (size_t)bh * NSEQ * DH;
  const bf16* Bb = kvT + (size_t)bh * DH * DH;

  f32x8 acc[2][2];
  acc[0][0] = zero8(); acc[0][1] = zero8();
  acc[1][0] = zero8(); acc[1][1] = zero8();

#pragma unroll
  for (int k0 = 0; k0 < DH; k0 += 32) {
    bf16x16 a0 = frag_ld(&Ab[(size_t)(rowBase + waveM * 32 + idx) * DH + k0], half);
    bf16x16 a1 = frag_ld(&Ab[(size_t)(rowBase + waveM * 32 + 16 + idx) * DH + k0], half);
    bf16x16 b0 = frag_ld(&Bb[(size_t)(waveN * 32 + idx) * DH + k0], half);
    bf16x16 b1 = frag_ld(&Bb[(size_t)(waveN * 32 + 16 + idx) * DH + k0], half);
    acc[0][0] = wmma_bf16(a0, b0, acc[0][0]);
    acc[0][1] = wmma_bf16(a0, b1, acc[0][1]);
    acc[1][0] = wmma_bf16(a1, b0, acc[1][0]);
    acc[1][1] = wmma_bf16(a1, b1, acc[1][1]);
  }

#pragma unroll
  for (int i = 0; i < 2; ++i)
#pragma unroll
    for (int jj = 0; jj < 2; ++jj) {
      const int m = waveN * 32 + jj * 16 + idx;
#pragma unroll
      for (int g = 0; g < 8; ++g) {
        const int n = rowBase + waveM * 32 + i * 16 + half * 8 + g;
        const float dv = den[(size_t)bh * NSEQ + n];
        attn[((size_t)bh * NSEQ + n) * DH + m] = acc[i][jj][g] / dv;
      }
    }
}

// ---------------------------------------------------------------------------
// LayerNorm over hidden (1024) + gate:  gated = LN(attn) * (h + beta)
// One block per (b,n) row; attn gathered from per-head planes (j = m*16 + h).
// ---------------------------------------------------------------------------
__global__ __launch_bounds__(256) void ln_gate(const float* __restrict__ attn,
                                               const float* __restrict__ hproj,
                                               const float* __restrict__ beta,
                                               const float* __restrict__ lnw,
                                               const float* __restrict__ lnb,
                                               bf16* __restrict__ gated) {
  __shared__ float red[256];
  const int row = blockIdx.x;       // 0..32767
  const int b = row >> 12, n = row & 4095;
  const int t = threadIdx.x;

  float vals[4];
  float s = 0.f, s2 = 0.f;
#pragma unroll
  for (int u = 0; u < 4; ++u) {
    const int j = t + u * 256;
    const int hh = j & 15, mm = j >> 4;
    float v = attn[(((size_t)b * HEADS + hh) * NSEQ + n) * DH + mm];
    vals[u] = v;
    s += v;
    s2 += v * v;
  }
  red[t] = s;
  __syncthreads();
  for (int o = 128; o > 0; o >>= 1) {
    if (t < o) red[t] += red[t + o];
    __syncthreads();
  }
  const float mu = red[0] * (1.f / 1024.f);
  __syncthreads();
  red[t] = s2;
  __syncthreads();
  for (int o = 128; o > 0; o >>= 1) {
    if (t < o) red[t] += red[t + o];
    __syncthreads();
  }
  const float var = red[0] * (1.f / 1024.f) - mu * mu;
  const float rv = rsqrtf(var + 1e-5f);

#pragma unroll
  for (int u = 0; u < 4; ++u) {
    const int j = t + u * 256;
    float ln = (vals[u] - mu) * rv * lnw[j] + lnb[j];
    float g = ln * (hproj[(size_t)row * HID + j] + beta[j]);
    gated[(size_t)row * HID + j] = (bf16)g;
  }
}

// ---------------------------------------------------------------------------
// Launcher
// ---------------------------------------------------------------------------
extern "C" void kernel_launch(void* const* d_in, const int* in_sizes, int n_in,
                              void* d_out, int out_size, void* d_ws, size_t ws_size,
                              hipStream_t stream) {
  const float* x    = (const float*)d_in[0];
  const float* Wh   = (const float*)d_in[1];
  const float* bh_  = (const float*)d_in[2];
  const float* Wq   = (const float*)d_in[3];
  const float* Wk   = (const float*)d_in[5];
  const float* Wv   = (const float*)d_in[7];
  const float* beta = (const float*)d_in[9];
  const float* lnw  = (const float*)d_in[10];
  const float* lnb  = (const float*)d_in[11];
  const float* Wo   = (const float*)d_in[12];
  const float* bo   = (const float*)d_in[13];
  float* out = (float*)d_out;

  char* ws = (char*)d_ws;
  size_t off = 0;
  auto alloc = [&](size_t bytes) -> void* {
    void* p = ws + off;
    off += (bytes + 255) & ~(size_t)255;
    return p;
  };
  bf16*  xb     = (bf16*)alloc((size_t)ROWS * HID * 2);
  bf16*  WT     = (bf16*)alloc((size_t)5 * HID * HID * 2);   // Wh,Wq,Wk,Wv,Wo
  float* hws    = (float*)alloc((size_t)ROWS * HID * 4);
  bf16*  qH     = (bf16*)alloc((size_t)BH * NSEQ * DH * 2);  // [bh][n][d]
  bf16*  kH     = (bf16*)alloc((size_t)BH * DH * NSEQ * 2);  // [bh][d][n]
  bf16*  vH     = (bf16*)alloc((size_t)BH * DH * NSEQ * 2);  // [bh][d][n]
  float* kvPart = (float*)alloc((size_t)BH * 8 * DH * DH * 4);
  bf16*  kvT    = (bf16*)alloc((size_t)BH * DH * DH * 2);    // [bh][m][d]
  float* ksum   = (float*)alloc((size_t)BH * DH * 4);
  float* den    = (float*)alloc((size_t)BH * NSEQ * 4);
  float* attn   = (float*)alloc((size_t)BH * NSEQ * DH * 4); // [bh][n][m]
  bf16*  gated  = (bf16*)alloc((size_t)ROWS * HID * 2);

  const size_t WSZ = (size_t)HID * HID;
  bf16* WTh = WT + 0 * WSZ;
  bf16* WTq = WT + 1 * WSZ;
  bf16* WTk = WT + 2 * WSZ;
  bf16* WTv = WT + 3 * WSZ;
  bf16* WTo = WT + 4 * WSZ;

  // Prep: weight transpose/convert + x convert
  transpose_w<<<dim3(32, 32), 256, 0, stream>>>(Wh, WTh);
  transpose_w<<<dim3(32, 32), 256, 0, stream>>>(Wq, WTq);
  transpose_w<<<dim3(32, 32), 256, 0, stream>>>(Wk, WTk);
  transpose_w<<<dim3(32, 32), 256, 0, stream>>>(Wv, WTv);
  transpose_w<<<dim3(32, 32), 256, 0, stream>>>(Wo, WTo);
  cvt_x<<<(ROWS * HID) / 256, 256, 0, stream>>>(x, xb);

  // Projections (x is 64MB bf16 -> L2 resident across the 4 launches)
  dim3 ggrid(ROWS / 128, HID / 64);
  gemm1024<0><<<ggrid, 256, 0, stream>>>(xb, WTh, bh_, hws, nullptr);
  gemm1024<1><<<ggrid, 256, 0, stream>>>(xb, WTq, nullptr, nullptr, qH);
  gemm1024<2><<<ggrid, 256, 0, stream>>>(xb, WTk, nullptr, nullptr, kH);
  gemm1024<3><<<ggrid, 256, 0, stream>>>(xb, WTv, nullptr, nullptr, vH);

  // Linear attention state
  ksum_kernel<<<BH * DH, 128, 0, stream>>>(kH, ksum);
  den_kernel<<<dim3(NSEQ / 256, BH), 256, 0, stream>>>(qH, ksum, den);
  kv_gemm<<<dim3(8, BH), 128, 0, stream>>>(vH, kH, kvPart);
  kv_reduce<<<(BH * DH * DH) / 256, 256, 0, stream>>>(kvPart, kvT);
  num_gemm<<<dim3(NSEQ / 128, BH), 256, 0, stream>>>(qH, kvT, den, attn);

  // LN + gate, then output projection with relu
  ln_gate<<<ROWS, 256, 0, stream>>>(attn, hws, beta, lnw, lnb, gated);
  gemm1024<4><<<ggrid, 256, 0, stream>>>(gated, WTo, bo, out, nullptr);
}
